// RGBD_PAM_Module_5729486373126
// MI455X (gfx1250) — matrix operand order repaired and checked by reading, not compiled, not run on hardware
//
#include <hip/hip_runtime.h>
#include <hip/hip_bf16.h>

// MI455X / gfx1250: wave32, WMMA 16x16x32 bf16 with f32 accumulate.
typedef __attribute__((ext_vector_type(16))) __bf16 bf16x16;
typedef __attribute__((ext_vector_type(8)))  __bf16 bf16x8;
typedef __attribute__((ext_vector_type(8)))  float  v8f;

#define NN 4096   // spatial positions H*W
#define CC 512    // channels
#define DD 128    // fused q/k depth (2 * C/8)
#define BB 4      // batch

__device__ inline v8f wmma_bf16(bf16x16 a, bf16x16 b, v8f c) {
  // 8 args: (neg_a, A, neg_b, B, c_mod, C, reuse_a, reuse_b)
  return __builtin_amdgcn_wmma_f32_16x16x32_bf16(false, a, false, b, (short)0, c,
                                                 false, false);
}

// A-fragment lane layout (16x32 bf16): per lane, two contiguous 8-element
// chunks at K = 8*hi and K = 16 + 8*hi.
__device__ inline bf16x16 load_a_frag(const __bf16* base, int hi) {
  bf16x8 lo = *(const bf16x8*)(base + 8 * hi);
  bf16x8 hi8 = *(const bf16x8*)(base + 16 + 8 * hi);
  return __builtin_shufflevector(lo, hi8, 0, 1, 2, 3, 4, 5, 6, 7,
                                 8, 9, 10, 11, 12, 13, 14, 15);
}

// ---------------------------------------------------------------------------
// Projection GEMM: Y[m,n] = sum_c W[m,c] * X[b,c,n] + bias[m]
// A-frag = W tile (16x32), B-frag = X tile (32x16). One 16x16 tile per wave.
// transposed=1 -> Y stored [b][n][DD] at row_off (Q^T / K^T). 0 -> [b][M][NN].
// ---------------------------------------------------------------------------
__global__ __launch_bounds__(256)
void proj_gemm(const float* __restrict__ W, const float* __restrict__ bias,
               const float* __restrict__ X, __bf16* __restrict__ Y,
               int M, int transposed, int row_off)
{
  const int tid = threadIdx.x;
  const int l  = tid & 31, w = tid >> 5;
  const int hi = l >> 4,  ln = l & 15;
  const int b  = blockIdx.z;
  const int mt = blockIdx.y * 16;
  const int n  = blockIdx.x * 128 + w * 16 + ln;
  const float* Xb = X + (size_t)b * CC * NN;
  const float* Wrow = W + (size_t)(mt + ln) * CC;

  v8f acc = {};
  for (int kc = 0; kc < CC; kc += 32) {
    bf16x16 a, bb;
    // A: two contiguous 32-byte f32 chunks per lane, converted to bf16
    float4 w0 = *(const float4*)(Wrow + kc + 8 * hi);
    float4 w1 = *(const float4*)(Wrow + kc + 8 * hi + 4);
    float4 w2 = *(const float4*)(Wrow + kc + 16 + 8 * hi);
    float4 w3 = *(const float4*)(Wrow + kc + 16 + 8 * hi + 4);
    a[0] = (__bf16)w0.x; a[1] = (__bf16)w0.y; a[2] = (__bf16)w0.z; a[3] = (__bf16)w0.w;
    a[4] = (__bf16)w1.x; a[5] = (__bf16)w1.y; a[6] = (__bf16)w1.z; a[7] = (__bf16)w1.w;
    a[8]  = (__bf16)w2.x; a[9]  = (__bf16)w2.y; a[10] = (__bf16)w2.z; a[11] = (__bf16)w2.w;
    a[12] = (__bf16)w3.x; a[13] = (__bf16)w3.y; a[14] = (__bf16)w3.z; a[15] = (__bf16)w3.w;
    // B: X[c,n] is strided in c -> scalar loads (coalesced across lanes in n)
#pragma unroll
    for (int v = 0; v < 8; ++v) {
      int kb = 2 * v + 8 * hi + (v >= 4 ? 8 : 0);
      bb[2 * v]     = (__bf16)Xb[(size_t)(kc + kb) * NN + n];
      bb[2 * v + 1] = (__bf16)Xb[(size_t)(kc + kb + 1) * NN + n];
    }
    acc = wmma_bf16(a, bb, acc);
  }
  if (transposed) {
    // rows r+8*hi are consecutive in the depth dim -> one 16-byte store
    bf16x8 yv;
#pragma unroll
    for (int r = 0; r < 8; ++r)
      yv[r] = (__bf16)(acc[r] + bias[mt + r + 8 * hi]);
    *(bf16x8*)(Y + ((size_t)b * NN + n) * DD + row_off + mt + 8 * hi) = yv;
  } else {
#pragma unroll
    for (int r = 0; r < 8; ++r) {
      int mrow = mt + r + 8 * hi;
      Y[((size_t)b * M + mrow) * NN + n] = (__bf16)(acc[r] + bias[mrow]);
    }
  }
}

// ---------------------------------------------------------------------------
// Energy tile: S(16q x 16j) = sum_{c=0..127} Q^T[i,c] * K[c,j], 4 WMMA (K=128)
// B-fragment is one contiguous 32-byte block per lane at K = 16*hi.
// ---------------------------------------------------------------------------
__device__ inline v8f energy_tile(const __bf16* __restrict__ Ktb, int jt,
                                  const bf16x16* qa, int hi, int ln)
{
  v8f s = {};
  const __bf16* kp = Ktb + (size_t)(jt * 16 + ln) * DD;
#pragma unroll
  for (int kc = 0; kc < 4; ++kc) {
    bf16x16 bb = *(const bf16x16*)(kp + kc * 32 + 16 * hi);
    s = wmma_bf16(qa[kc], bb, s);
  }
  return s;
}

// ---------------------------------------------------------------------------
// Flash-style attention: one workgroup (8 waves) per (batch, 16-query tile).
// ---------------------------------------------------------------------------
__global__ __launch_bounds__(256)
void attn_flash(const __bf16* __restrict__ Qt, const __bf16* __restrict__ Kt,
                const __bf16* __restrict__ Vb, const float* __restrict__ x_rgb,
                const float* __restrict__ gamma_p, float* __restrict__ out)
{
  __shared__ __align__(32) __bf16 Pbuf[16 * 256];   // 8 KB staged probabilities
  __shared__ float stat_m[8][16];
  __shared__ float stat_l[8][16];
  __shared__ float fin_m[16];
  __shared__ float fin_il[16];

  const int tid = threadIdx.x;
  const int l  = tid & 31, w = tid >> 5;
  const int hi = l >> 4,  ln = l & 15;
  const int b = blockIdx.y;
  const int itile = blockIdx.x * 16;

  const __bf16* Qb  = Qt + (size_t)b * NN * DD;
  const __bf16* Ktb = Kt + (size_t)b * NN * DD;
  const __bf16* Vbb = Vb + (size_t)b * CC * NN;

  // Q^T A-fragments (16 rows x 128 depth), resident for both sweeps.
  bf16x16 qa[4];
  {
    const __bf16* qp = Qb + (size_t)(itile + ln) * DD;
#pragma unroll
    for (int kc = 0; kc < 4; ++kc)
      qa[kc] = *(const bf16x16*)(qp + kc * 32 + 16 * hi);
  }

  // ---- Sweep 1: per-row (max, sum-exp), waves stride over j-tiles ----
  float m_r[8], l_r[8];
#pragma unroll
  for (int r = 0; r < 8; ++r) { m_r[r] = -3.0e38f; l_r[r] = 0.0f; }

  for (int jt = w; jt < NN / 16; jt += 8) {
    // speculative prefetch of the next j-tile's K rows (gfx1250 global_prefetch)
    __builtin_prefetch(Ktb + (size_t)((jt + 8) * 16 + ln) * DD, 0, 1);
    v8f s = energy_tile(Ktb, jt, qa, hi, ln);
#pragma unroll
    for (int r = 0; r < 8; ++r) {
      float sv = s[r];
      float tmax = sv;                              // max over 16-lane row group
      tmax = fmaxf(tmax, __shfl_xor(tmax, 1));
      tmax = fmaxf(tmax, __shfl_xor(tmax, 2));
      tmax = fmaxf(tmax, __shfl_xor(tmax, 4));
      tmax = fmaxf(tmax, __shfl_xor(tmax, 8));
      float newm = fmaxf(m_r[r], tmax);
      float pe = __expf(sv - newm);
      pe += __shfl_xor(pe, 1);
      pe += __shfl_xor(pe, 2);
      pe += __shfl_xor(pe, 4);
      pe += __shfl_xor(pe, 8);
      l_r[r] = l_r[r] * __expf(m_r[r] - newm) + pe;
      m_r[r] = newm;
    }
  }
  if (ln == 0) {
#pragma unroll
    for (int r = 0; r < 8; ++r) {
      stat_m[w][r + 8 * hi] = m_r[r];
      stat_l[w][r + 8 * hi] = l_r[r];
    }
  }
  __syncthreads();
  if (tid < 16) {
    float M = -3.0e38f;
    for (int ww = 0; ww < 8; ++ww) M = fmaxf(M, stat_m[ww][tid]);
    float L = 0.0f;
    for (int ww = 0; ww < 8; ++ww) L += stat_l[ww][tid] * __expf(stat_m[ww][tid] - M);
    fin_m[tid]  = M;
    fin_il[tid] = 1.0f / L;
  }
  __syncthreads();
  float mrow[8], ilrow[8];
#pragma unroll
  for (int r = 0; r < 8; ++r) { mrow[r] = fin_m[r + 8 * hi]; ilrow[r] = fin_il[r + 8 * hi]; }

  // ---- Sweep 2: O += softmax(S) * V^T, 256-column chunks ----
  v8f o[4] = {};
  for (int chunk = 0; chunk < 16; ++chunk) {
    const int jbase = chunk * 256;
    // Phase A: each wave produces 2 probability tiles into LDS
#pragma unroll
    for (int t = 0; t < 2; ++t) {
      int jt = chunk * 16 + w * 2 + t;
      v8f s = energy_tile(Ktb, jt, qa, hi, ln);
      int jloc = (w * 2 + t) * 16 + ln;
#pragma unroll
      for (int r = 0; r < 8; ++r) {
        float p = __expf(s[r] - mrow[r]) * ilrow[r];
        Pbuf[(r + 8 * hi) * 256 + jloc] = (__bf16)p;
      }
    }
    __syncthreads();
    // Phase B: every wave multiplies shared P against its 4 c-tiles of V
    for (int kc = 0; kc < 8; ++kc) {
      bf16x16 af = *(const bf16x16*)&Pbuf[ln * 256 + kc * 32 + 16 * hi];  // one 32-byte LDS load
#pragma unroll
      for (int t = 0; t < 4; ++t) {
        int c = (w * 4 + t) * 16 + ln;
        // V B-fragment: one contiguous 32-byte block per lane
        bf16x16 bb = *(const bf16x16*)(Vbb + (size_t)c * NN + jbase + kc * 32 + 16 * hi);
        o[t] = wmma_bf16(af, bb, o[t]);
      }
    }
    __syncthreads();
  }

  // Epilogue: out = gamma * O + x_rgb (rows r+8*hi are consecutive in i)
  const float g = gamma_p[0];
#pragma unroll
  for (int t = 0; t < 4; ++t) {
    int c = (w * 4 + t) * 16 + ln;
    size_t base = ((size_t)b * CC + c) * NN + itile + 8 * hi;
    float4 x0 = *(const float4*)(x_rgb + base);
    float4 x1 = *(const float4*)(x_rgb + base + 4);
    float4 r0, r1;
    r0.x = g * o[t][0] + x0.x; r0.y = g * o[t][1] + x0.y;
    r0.z = g * o[t][2] + x0.z; r0.w = g * o[t][3] + x0.w;
    r1.x = g * o[t][4] + x1.x; r1.y = g * o[t][5] + x1.y;
    r1.z = g * o[t][6] + x1.z; r1.w = g * o[t][7] + x1.w;
    *(float4*)(out + base)     = r0;
    *(float4*)(out + base + 4) = r1;
  }
}

extern "C" void kernel_launch(void* const* d_in, const int* in_sizes, int n_in,
                              void* d_out, int out_size, void* d_ws, size_t ws_size,
                              hipStream_t stream) {
  (void)in_sizes; (void)n_in; (void)out_size; (void)ws_size;
  const float* x_rgb = (const float*)d_in[0];
  const float* x_dep = (const float*)d_in[1];
  const float* Wq  = (const float*)d_in[2];  const float* bq  = (const float*)d_in[3];
  const float* Wk  = (const float*)d_in[4];  const float* bk  = (const float*)d_in[5];
  const float* Wdq = (const float*)d_in[6];  const float* bdq = (const float*)d_in[7];
  const float* Wdk = (const float*)d_in[8];  const float* bdk = (const float*)d_in[9];
  const float* Wv  = (const float*)d_in[10]; const float* bv  = (const float*)d_in[11];
  const float* gamma = (const float*)d_in[12];
  float* out = (float*)d_out;

  char* ws = (char*)d_ws;
  const size_t qk_bytes = (size_t)BB * NN * DD * sizeof(__bf16);   // 4 MB each
  __bf16* Qt = (__bf16*)ws;                       // [B][N][128] bf16
  __bf16* Kt = (__bf16*)(ws + qk_bytes);          // [B][N][128] bf16
  __bf16* Vb = (__bf16*)(ws + 2 * qk_bytes);      // [B][512][N] bf16 (16 MB)

  dim3 blk(256);
  // Fused projections -> Q^T / K^T (rgb rows 0..63, depth rows 64..127) and V.
  proj_gemm<<<dim3(NN / 128,  4, BB), blk, 0, stream>>>(Wq,  bq,  x_rgb, Qt,  64, 1,  0);
  proj_gemm<<<dim3(NN / 128,  4, BB), blk, 0, stream>>>(Wdq, bdq, x_dep, Qt,  64, 1, 64);
  proj_gemm<<<dim3(NN / 128,  4, BB), blk, 0, stream>>>(Wk,  bk,  x_rgb, Kt,  64, 1,  0);
  proj_gemm<<<dim3(NN / 128,  4, BB), blk, 0, stream>>>(Wdk, bdk, x_dep, Kt,  64, 1, 64);
  proj_gemm<<<dim3(NN / 128, 32, BB), blk, 0, stream>>>(Wv,  bv,  x_rgb, Vb, 512, 0,  0);
  // Fused energy + softmax + P*V^T + residual.
  attn_flash<<<dim3(NN / 16, BB), blk, 0, stream>>>(Qt, Kt, Vb, x_rgb, gamma, out);
}
